// SpectralInterpolate_19963007992453
// MI455X (gfx1250) — compile-verified, matrix operand order checked by reading, not run on hardware
//
#include <hip/hip_runtime.h>
#include <stdint.h>

typedef __attribute__((ext_vector_type(16))) _Float16 v16h;
typedef __attribute__((ext_vector_type(8)))  float    v8f;

#define B_ 4
#define C_ 512
#define T_ 1024
#define F_ 4096                        // floor(MAXFREQ*T/(2*timespan)/freqstep) = 4*T
#define TWO_PI 6.28318530717958647692f
#define FREQSTEP (1.0f / 8192.0f)      // 1/(timespan*OSFREQ), timespan = T = 1024

// K index inside a 16x32 f16 WMMA fragment (wave32 layout, ISA 7.12.2):
// lanes 0-15 : elems 0-7 -> K 0-7,  elems 8-15 -> K 16-23
// lanes 16-31: elems 0-7 -> K 8-15, elems 8-15 -> K 24-31
__device__ __forceinline__ int kmap(int e, int half) {
    return e + (e >= 8 ? 8 : 0) + 8 * half;
}

// ---------------------------------------------------------------------------
// Kernel 0: tau[b,f] = atan2(sum w*sin(2*ang*t), sum w*cos(2*ang*t)) / (2*ang)
//           inv_den_s/c[b,f] = 1 / sum_t w * sin/cos(ang*(t - tau))^2
// one block per (b,f)
// ---------------------------------------------------------------------------
__global__ void tau_den_kernel(const uint8_t* __restrict__ mask,
                               float* __restrict__ tau,
                               float* __restrict__ inv_den_s,
                               float* __restrict__ inv_den_c) {
    const int bf = blockIdx.x;
    const int b  = bf / F_;
    const int f  = bf % F_;
    const float ang = TWO_PI * FREQSTEP * (float)(f + 1);
    const int tid = threadIdx.x;

    __shared__ float r0[256];
    __shared__ float r1[256];
    __shared__ float s_tau;

    float ns = 0.f, nc = 0.f;
    for (int j = 0; j < T_ / 256; ++j) {
        int t = tid + j * 256;
        float w = (float)mask[b * T_ + t];
        float s, c;
        __sincosf(2.0f * ang * (float)(t + 1), &s, &c);
        ns += w * s; nc += w * c;
    }
    r0[tid] = ns; r1[tid] = nc;
    __syncthreads();
    for (int off = 128; off > 0; off >>= 1) {
        if (tid < off) { r0[tid] += r0[tid + off]; r1[tid] += r1[tid + off]; }
        __syncthreads();
    }
    if (tid == 0) {
        float tv = atan2f(r0[0], r1[0]) / (2.0f * ang);
        s_tau = tv;
        tau[bf] = tv;
    }
    __syncthreads();
    const float tv = s_tau;

    float ds = 0.f, dc = 0.f;
    for (int j = 0; j < T_ / 256; ++j) {
        int t = tid + j * 256;
        float w = (float)mask[b * T_ + t];
        float s, c;
        __sincosf(ang * ((float)(t + 1) - tv), &s, &c);
        ds += w * s * s; dc += w * c * c;
    }
    r0[tid] = ds; r1[tid] = dc;
    __syncthreads();
    for (int off = 128; off > 0; off >>= 1) {
        if (tid < off) { r0[tid] += r0[tid + off]; r1[tid] += r1[tid + off]; }
        __syncthreads();
    }
    if (tid == 0) { inv_den_s[bf] = 1.0f / r0[0]; inv_den_c[bf] = 1.0f / r1[0]; }
}

// ---------------------------------------------------------------------------
// Kernel 1 (fit GEMM): spec[b,f,c] = (sum_t basis_masked[f,t]*data[c,t]) * inv_den
// block: (b, 16-freq tile) covers ALL 512 channels: 8 waves x 4 tiles of 16.
// Masked basis chunk (16 x 512) generated once per block -> shared by 32 tiles.
// ---------------------------------------------------------------------------
#define KC1 512
#define PAD1 8   // LDS row stride 520 halves = 260 dwords -> rows hit distinct banks
#define NT1 4    // channel tiles per wave

__global__ void fit_kernel(const float* __restrict__ data,
                           const uint8_t* __restrict__ mask,
                           const float* __restrict__ tau,
                           const float* __restrict__ inv_den_s,
                           const float* __restrict__ inv_den_c,
                           float* __restrict__ sin_spec,
                           float* __restrict__ cos_spec) {
    const int b    = blockIdx.z;
    const int f0   = blockIdx.y * 16;
    const int tid  = threadIdx.x;
    const int wave = tid >> 5;
    const int lane = tid & 31;
    const int half = lane >> 4;
    const int mrow = lane & 15;
    const int c0   = wave * (16 * NT1);          // wave covers [c0, c0+64)

    __shared__ _Float16 sinb[16][KC1 + PAD1];
    __shared__ _Float16 cosb[16][KC1 + PAD1];
    __shared__ float    s_tau[16];
    if (tid < 16) s_tau[tid] = tau[b * F_ + f0 + tid];
    __syncthreads();

    v8f accS[NT1], accC[NT1];
    #pragma unroll
    for (int ct = 0; ct < NT1; ++ct) {
        accS[ct] = (v8f){0.f,0.f,0.f,0.f,0.f,0.f,0.f,0.f};
        accC[ct] = (v8f){0.f,0.f,0.f,0.f,0.f,0.f,0.f,0.f};
    }

    for (int kc = 0; kc < T_; kc += KC1) {
        // cooperative masked-basis generation (consecutive t per warp -> coalesced)
        for (int j = 0; j < (16 * KC1) / 256; ++j) {
            int idx = tid + j * 256;
            int fi  = idx / KC1;
            int ti  = idx % KC1;
            int t   = kc + ti;
            float ang = TWO_PI * FREQSTEP * (float)(f0 + fi + 1);
            float w   = (float)mask[b * T_ + t];
            float s, c;
            __sincosf(ang * ((float)(t + 1) - s_tau[fi]), &s, &c);
            sinb[fi][ti] = (_Float16)(s * w);
            cosb[fi][ti] = (_Float16)(c * w);
        }
        __syncthreads();

        // prefetch next chunk of this wave's data rows into L2/L0
        if (kc + KC1 < T_) {
            __builtin_prefetch(&data[((size_t)b * C_ + (c0 + mrow)) * T_ + (kc + KC1) + lane], 0, 3);
        }

        for (int kk = 0; kk < KC1; kk += 32) {
            v16h a_s, a_c;
            #pragma unroll
            for (int e = 0; e < 16; ++e) {
                int k = kk + kmap(e, half);
                a_s[e] = sinb[mrow][k];
                a_c[e] = cosb[mrow][k];
            }
            #pragma unroll
            for (int ct = 0; ct < NT1; ++ct) {
                v16h bb;
                #pragma unroll
                for (int e = 0; e < 16; ++e) {
                    int k = kk + kmap(e, half);
                    bb[e] = (_Float16)data[((size_t)b * C_ + (c0 + ct * 16 + mrow)) * T_ + (kc + k)];
                }
                accS[ct] = __builtin_amdgcn_wmma_f32_16x16x32_f16(false, a_s, false, bb,
                                                                  (short)0, accS[ct], false, false);
                accC[ct] = __builtin_amdgcn_wmma_f32_16x16x32_f16(false, a_c, false, bb,
                                                                  (short)0, accC[ct], false, false);
            }
        }
        __syncthreads();
    }

    const int n = lane & 15;
    #pragma unroll
    for (int ct = 0; ct < NT1; ++ct) {
        #pragma unroll
        for (int r = 0; r < 8; ++r) {
            int f = f0 + r + 8 * half;
            size_t o = ((size_t)b * F_ + f) * C_ + (c0 + ct * 16 + n);
            sin_spec[o] = accS[ct][r] * inv_den_s[b * F_ + f];
            cos_spec[o] = accC[ct][r] * inv_den_c[b * F_ + f];
        }
    }
}

// ---------------------------------------------------------------------------
// Kernel 2 (recon GEMM): recon[b,c,t] = sum_f sinb[f,t]*sspec[f,c] + cosb*cspec
// block: (b, 16-t tile, half of C); 8 waves x 2 channel tiles.
// Unmasked basis chunk (512 x 16) in LDS, reused across the 16 tiles.
// ---------------------------------------------------------------------------
#define KC2 512
#define NT2 2    // channel tiles per wave

__global__ void recon_kernel(const float* __restrict__ tau,
                             const float* __restrict__ sin_spec,
                             const float* __restrict__ cos_spec,
                             float* __restrict__ recon) {
    const int b    = blockIdx.z;
    const int t0   = blockIdx.y * 16;
    const int tid  = threadIdx.x;
    const int wave = tid >> 5;
    const int lane = tid & 31;
    const int half = lane >> 4;
    const int mrow = lane & 15;
    const int c0   = (blockIdx.x * 8 + wave) * (16 * NT2);

    __shared__ _Float16 sb[KC2][16];
    __shared__ _Float16 cb[KC2][16];

    v8f acc[NT2];
    #pragma unroll
    for (int ct = 0; ct < NT2; ++ct)
        acc[ct] = (v8f){0.f,0.f,0.f,0.f,0.f,0.f,0.f,0.f};

    for (int fc = 0; fc < F_; fc += KC2) {
        // unmasked basis chunk (f-major, t contiguous)
        for (int j = 0; j < (KC2 * 16) / 256; ++j) {
            int idx = tid * ((KC2 * 16) / 256) + j;
            int fi  = idx / 16;
            int ti  = idx % 16;
            int f   = fc + fi;
            float ang = TWO_PI * FREQSTEP * (float)(f + 1);
            float s, c;
            __sincosf(ang * ((float)(t0 + ti + 1) - tau[b * F_ + f]), &s, &c);
            sb[fi][ti] = (_Float16)s;
            cb[fi][ti] = (_Float16)c;
        }
        __syncthreads();

        for (int kk = 0; kk < KC2; kk += 32) {
            v16h b_s, b_c;
            #pragma unroll
            for (int e = 0; e < 16; ++e) {
                int k = kmap(e, half);
                b_s[e] = sb[kk + k][mrow];
                b_c[e] = cb[kk + k][mrow];
            }
            #pragma unroll
            for (int ct = 0; ct < NT2; ++ct) {
                v16h a_s, a_c;
                #pragma unroll
                for (int e = 0; e < 16; ++e) {
                    int k = kmap(e, half);
                    size_t o = ((size_t)b * F_ + (fc + kk + k)) * C_ + (c0 + ct * 16 + mrow);
                    a_s[e] = (_Float16)sin_spec[o];   // L2-resident (67MB < 192MB L2)
                    a_c[e] = (_Float16)cos_spec[o];
                }
                acc[ct] = __builtin_amdgcn_wmma_f32_16x16x32_f16(false, a_s, false, b_s,
                                                                 (short)0, acc[ct], false, false);
                acc[ct] = __builtin_amdgcn_wmma_f32_16x16x32_f16(false, a_c, false, b_c,
                                                                 (short)0, acc[ct], false, false);
            }
        }
        __syncthreads();
    }

    const int n = lane & 15;
    #pragma unroll
    for (int ct = 0; ct < NT2; ++ct) {
        #pragma unroll
        for (int r = 0; r < 8; ++r) {
            int c = c0 + ct * 16 + r + 8 * half;
            recon[((size_t)b * C_ + c) * T_ + t0 + n] = acc[ct][r];
        }
    }
}

// ---------------------------------------------------------------------------
// Kernel 3: per-(b,c) weighted-std normalization + mask select, in place on out
// ---------------------------------------------------------------------------
__global__ void finalize_kernel(const float* __restrict__ data,
                                const uint8_t* __restrict__ mask,
                                float* __restrict__ out) {
    const int bc = blockIdx.x;
    const int b  = bc / C_;
    const int tid = threadIdx.x;

    const float*   drow = data + (size_t)bc * T_;
    float*         rrow = out  + (size_t)bc * T_;
    const uint8_t* wrow = mask + (size_t)b * T_;

    float s1r = 0.f, s2r = 0.f, s1d = 0.f, s2d = 0.f, cnt = 0.f;
    for (int j = 0; j < T_ / 256; ++j) {
        int t = tid + j * 256;
        float w  = (float)wrow[t];
        float rv = rrow[t];
        float dv = drow[t];
        s1r += w * rv; s2r += w * rv * rv;
        s1d += w * dv; s2d += w * dv * dv;
        cnt += w;
    }
    __shared__ float red[5][256];
    red[0][tid] = s1r; red[1][tid] = s2r; red[2][tid] = s1d;
    red[3][tid] = s2d; red[4][tid] = cnt;
    __syncthreads();
    for (int off = 128; off > 0; off >>= 1) {
        if (tid < off) {
            #pragma unroll
            for (int q = 0; q < 5; ++q) red[q][tid] += red[q][tid + off];
        }
        __syncthreads();
    }
    const float n  = red[4][0];
    const float vr = (red[1][0] - red[0][0] * red[0][0] / n) / (n - 1.0f);
    const float vd = (red[3][0] - red[2][0] * red[2][0] / n) / (n - 1.0f);
    const float inv_norm = sqrtf(vd / vr);   // recon / (std_r/std_d)

    for (int j = 0; j < T_ / 256; ++j) {
        int t = tid + j * 256;
        float w = (float)wrow[t];
        rrow[t] = (w > 0.5f) ? drow[t] : rrow[t] * inv_norm;
    }
}

// ---------------------------------------------------------------------------
extern "C" void kernel_launch(void* const* d_in, const int* in_sizes, int n_in,
                              void* d_out, int out_size, void* d_ws, size_t ws_size,
                              hipStream_t stream) {
    (void)in_sizes; (void)n_in; (void)out_size; (void)ws_size;
    const float*   data = (const float*)d_in[0];
    const uint8_t* mask = (const uint8_t*)d_in[1];
    float* out = (float*)d_out;

    float* ws        = (float*)d_ws;
    float* tau       = ws;                               // B*F
    float* inv_den_s = tau + (size_t)B_ * F_;            // B*F
    float* inv_den_c = inv_den_s + (size_t)B_ * F_;      // B*F
    float* sin_spec  = inv_den_c + (size_t)B_ * F_;      // B*F*C
    float* cos_spec  = sin_spec + (size_t)B_ * F_ * C_;  // B*F*C

    tau_den_kernel<<<B_ * F_, 256, 0, stream>>>(mask, tau, inv_den_s, inv_den_c);

    dim3 g1(1, F_ / 16, B_);                 // 1024 blocks, each owns all of C
    fit_kernel<<<g1, 256, 0, stream>>>(data, mask, tau, inv_den_s, inv_den_c,
                                       sin_spec, cos_spec);

    dim3 g2(C_ / (16 * 8 * NT2), T_ / 16, B_);   // 512 blocks
    recon_kernel<<<g2, 256, 0, stream>>>(tau, sin_spec, cos_spec, out);

    finalize_kernel<<<B_ * C_, 256, 0, stream>>>(data, mask, out);
}
